// MambaFBAMAgent_14723147891397
// MI455X (gfx1250) — matrix operand order repaired and checked by reading, not compiled, not run on hardware
//
#include <hip/hip_runtime.h>
#include <cstdint>
#include <cstddef>

// ---------- problem constants (from reference) ----------
constexpr int kT    = 16;      // frames / batch
constexpr int kL    = 4800;    // H*W tokens per frame
constexpr int kBL   = kT * kL; // 76800 rows
constexpr int kDM   = 128;     // model dim
constexpr int kDI   = 256;     // inner dim
constexpr int kNS   = 16;      // mamba states
constexpr int kDTR  = 8;       // dt rank
constexpr int kKC   = 4;       // conv kernel
constexpr int kHID  = 256;     // lstm hidden
constexpr int kAEMB = 32;
constexpr int kNA   = 12;
constexpr int kNL   = 2;
constexpr int kXP   = 48;      // x_proj cols padded 40 -> 48
constexpr int kNC   = 40;      // scan chunks
constexpr int kCL   = kL / kNC; // 120 steps per chunk

typedef __attribute__((ext_vector_type(16))) __bf16 v16bf;
typedef __attribute__((ext_vector_type(8)))  float  v8f;
typedef __attribute__((ext_vector_type(4)))  float  f4;

struct alignas(8) bfx4 { __bf16 h[4]; };

#if __has_builtin(__builtin_amdgcn_sched_group_barrier)
#define SCHED_GROUP(mask, size, id) __builtin_amdgcn_sched_group_barrier((mask), (size), (id))
#else
#define SCHED_GROUP(mask, size, id)
#endif

__device__ __forceinline__ float sigf(float x) { return 1.f / (1.f + __expf(-x)); }
__device__ __forceinline__ float softplusf(float x) {
  return (x > 15.f) ? x : __logf(1.f + __expf(x));
}

// ---------- weight conversion ----------
__global__ void k_f2bf(const float* __restrict__ s, __bf16* __restrict__ d, int n) {
  int i = blockIdx.x * 256 + threadIdx.x;
  if (i < n) d[i] = (__bf16)s[i];
}

// x_proj_w [NL,40,DI] -> padded bf16 [NL,48,DI]
__global__ void k_xpw_pad(const float* __restrict__ s, __bf16* __restrict__ d) {
  int i = blockIdx.x * 256 + threadIdx.x; // over kNL*48*kDI
  int layer = i / (kXP * kDI);
  int r     = (i / kDI) % kXP;
  int c     = i % kDI;
  float v = (r < 40) ? s[(size_t)layer * 40 * kDI + r * kDI + c] : 0.f;
  d[i] = (__bf16)v;
}

// ---------- token embedding + positional ----------
__global__ void k_embed(const int* __restrict__ frames, const float* __restrict__ emb,
                        const float* __restrict__ pos, float* __restrict__ x) {
  int row = blockIdx.x;          // b*L + l
  int t   = threadIdx.x;         // 0..127
  int tok = frames[row];
  int l   = row % kL;
  x[(size_t)row * kDM + t] = emb[(size_t)tok * kDM + t] + pos[(size_t)l * kDM + t];
}

// ---------- layernorm (per-wave row, bf16 output for WMMA A operand) ----------
__global__ void k_layernorm_bf16(const float* __restrict__ x, const float* __restrict__ w,
                                 const float* __restrict__ b, __bf16* __restrict__ out) {
  int wave = threadIdx.x >> 5, lane = threadIdx.x & 31;
  int row  = blockIdx.x * 8 + wave;
  f4 v = ((const f4*)(x + (size_t)row * kDM))[lane];
  float s  = v[0] + v[1] + v[2] + v[3];
  float s2 = v[0]*v[0] + v[1]*v[1] + v[2]*v[2] + v[3]*v[3];
  #pragma unroll
  for (int o = 16; o > 0; o >>= 1) { s += __shfl_xor(s, o); s2 += __shfl_xor(s2, o); }
  float m   = s  * (1.f / kDM);
  float inv = rsqrtf(s2 * (1.f / kDM) - m * m + 1e-5f);
  f4 wv = ((const f4*)w)[lane];
  f4 bv = ((const f4*)b)[lane];
  bfx4 o;
  #pragma unroll
  for (int e = 0; e < 4; e++) o.h[e] = (__bf16)((v[e] - m) * inv * wv[e] + bv[e]);
  *(bfx4*)(out + (size_t)row * kDM + lane * 4) = o;
}

// ---------- bf16 WMMA GEMM: C[M,N] (+)= A[M,K] * W[N,K]^T ----------
// one wave -> 16 x (16*NT) tile; compile-time K; register double-buffered K-loop.
// sched_group_barrier enforces, per chunk: [2*(NT+1) VMEM reads (next chunk)] then
// [NT WMMAs (current chunk)], so loads overlap WMMAs and RA must keep
// independent staging buffers -> partial s_wait_loadcnt instead of wait-0.
template <int NT, int K, bool ACC>
__global__ __launch_bounds__(256, 1)
void k_gemm_bf16(const __bf16* __restrict__ A, const __bf16* __restrict__ W,
                 float* __restrict__ C, int Mtiles, int GroupsN, int ldc) {
  int wg = blockIdx.x * 8 + (threadIdx.x >> 5);
  if (wg >= Mtiles * GroupsN) return;          // wave-uniform; EXEC stays all-1s for WMMA
  int lane = threadIdx.x & 31;
  int m0 = (wg / GroupsN) * 16;
  int n0 = (wg % GroupsN) * (16 * NT);
  int r  = lane & 15;
  int hf = lane >> 4;

  // A operand: lane<16 holds row r, k = {kc+0..7, kc+16..23}; lane>=16 shifts +8
  const __bf16* ap = A + (size_t)(m0 + r) * K + hf * 8;
  // B operand from W[N,K]: lane n holds col n, k = {kc+0..7, kc+16..23}; lane>=16 shifts +8
  const __bf16* bp[NT];
  #pragma unroll
  for (int t = 0; t < NT; t++) bp[t] = W + (size_t)(n0 + t * 16 + r) * K + hf * 8;

  union AU { f4 f[2]; v16bf v; };
  v8f acc[NT] = {};
  AU a0, b0[NT];
  a0.f[0] = *(const f4*)(ap);
  a0.f[1] = *(const f4*)(ap + 16);
  #pragma unroll
  for (int t = 0; t < NT; t++) {
    b0[t].f[0] = *(const f4*)(bp[t]);
    b0[t].f[1] = *(const f4*)(bp[t] + 16);
  }
  #pragma unroll
  for (int kc = 0; kc < K; kc += 32) {
    AU a1, b1[NT];
    const int kn = kc + 32;
    if (kn < K) {                                   // compile-time after unroll
      a1.f[0] = *(const f4*)(ap + kn);
      a1.f[1] = *(const f4*)(ap + kn + 16);
      #pragma unroll
      for (int t = 0; t < NT; t++) {
        b1[t].f[0] = *(const f4*)(bp[t] + kn);
        b1[t].f[1] = *(const f4*)(bp[t] + kn + 16);
      }
    }
    #pragma unroll
    for (int t = 0; t < NT; t++)
      acc[t] = __builtin_amdgcn_wmma_f32_16x16x32_bf16(
                 false, a0.v, false, b0[t].v, (short)0, acc[t], false, false);
    // scheduling directive: issue next-chunk loads before this chunk's WMMAs
    if (kn < K) SCHED_GROUP(0x20, 2 * (NT + 1), 0);   // VMEM reads
    SCHED_GROUP(0x8, NT, 0);                          // WMMA group
    if (kn < K) {
      a0 = a1;
      #pragma unroll
      for (int t = 0; t < NT; t++) b0[t] = b1[t];
    }
  }
  // C/D layout: lane<16 -> N=lane, M=vgpr r ; lane>=16 -> N=lane-16, M=8+r
  #pragma unroll
  for (int t = 0; t < NT; t++) {
    float* cp = C + (size_t)(m0 + hf * 8) * ldc + n0 + t * 16 + r;
    #pragma unroll
    for (int rr = 0; rr < 8; rr++) {
      if (ACC) cp[(size_t)rr * ldc] += acc[t][rr];
      else     cp[(size_t)rr * ldc]  = acc[t][rr];
    }
  }
}

// ---------- depthwise causal conv (K=4) + SiLU, reads xz[:, 0:DI] f32, writes bf16 ----------
__global__ void k_conv_silu(const float* __restrict__ xz, const float* __restrict__ cw,
                            const float* __restrict__ cb, __bf16* __restrict__ xi) {
  int i   = blockIdx.x * 256 + threadIdx.x;   // over kBL*kDI
  int d   = i & (kDI - 1);
  int row = i >> 8;                           // b*L + l
  int l   = row % kL;
  f4 w = *(const f4*)(cw + d * kKC);
  float acc = cb[d];
  #pragma unroll
  for (int k = 0; k < kKC; k++) {
    int ls = l + k - (kKC - 1);
    if (ls >= 0) acc += w[k] * xz[(size_t)(row + k - (kKC - 1)) * (2 * kDI) + d];
  }
  float s = acc * sigf(acc);
  xi[(size_t)row * kDI + d] = (__bf16)s;
}

// ======================================================================
// Chunked parallel selective scan (3 passes).
// Recurrence h <- dA (.) h + dt*B*x is diagonal-linear => associative.
// Pass 1: per (b,chunk,d): P = prod(dA), S = zero-init end state  (5120 waves)
// Pass 2: per (b,d): sequential combine over 40 chunks -> Hst      (tiny)
// Pass 3: per (b,chunk,d): re-run chunk from Hst, emit gated y     (5120 waves)
// ======================================================================

constexpr int kPFD = 8;   // prefetch distance (rows ahead) for scan streams

__global__ void k_scan_part1(const float* __restrict__ dbl, const __bf16* __restrict__ xi,
                             const float* __restrict__ dtw, const float* __restrict__ dtb,
                             const float* __restrict__ alog,
                             float* __restrict__ P, float* __restrict__ S) {
  int bc = blockIdx.x;               // b*kNC + c
  int b = bc / kNC, c = bc % kNC;
  int d = threadIdx.x;
  float wdt[kDTR];
  #pragma unroll
  for (int e = 0; e < kDTR; e++) wdt[e] = dtw[d * kDTR + e];
  float Anr[kNS];
  #pragma unroll
  for (int n = 0; n < kNS; n++) Anr[n] = -__expf(alog[d * kNS + n]);
  float bdt = dtb[d];
  float h[kNS] = {};
  float p[kNS];
  #pragma unroll
  for (int n = 0; n < kNS; n++) p[n] = 1.f;

  size_t row0 = (size_t)b * kL + (size_t)c * kCL;
  for (int l = 0; l < kCL; l++) {
    size_t row = row0 + l;
    if (l + kPFD < kCL) {   // pull upcoming rows into cache (global_prefetch_b8)
      __builtin_prefetch(dbl + (row + kPFD) * kXP, 0, 3);
      __builtin_prefetch(xi + (row + kPFD) * kDI + d, 0, 3);
    }
    const f4* q = (const f4*)(dbl + row * kXP);
    f4 r0 = q[0], r1 = q[1];
    float dl = bdt;
    #pragma unroll
    for (int e = 0; e < 4; e++) { dl += r0[e] * wdt[e]; dl += r1[e] * wdt[4 + e]; }
    float dt = softplusf(dl);
    float dtxt = dt * (float)xi[row * kDI + d];
    #pragma unroll
    for (int n = 0; n < kNS; n++) {
      float Bn = q[2 + (n >> 2)][n & 3];           // dbl cols 8..23
      float dA = __expf(dt * Anr[n]);
      h[n] = dA * h[n] + dtxt * Bn;
      p[n] *= dA;
    }
  }
  float* Pp = P + ((size_t)bc * kDI + d) * kNS;
  float* Sp = S + ((size_t)bc * kDI + d) * kNS;
  #pragma unroll
  for (int n = 0; n < kNS; n++) { Pp[n] = p[n]; Sp[n] = h[n]; }
}

__global__ void k_scan_part2(const float* __restrict__ P, const float* __restrict__ S,
                             float* __restrict__ Hst) {
  int b = blockIdx.x;       // 0..15
  int d = threadIdx.x;      // 0..255
  float h[kNS] = {};
  for (int c = 0; c < kNC; c++) {
    size_t base = (((size_t)b * kNC + c) * kDI + d) * kNS;
    #pragma unroll
    for (int n = 0; n < kNS; n++) {
      Hst[base + n] = h[n];
      h[n] = P[base + n] * h[n] + S[base + n];
    }
  }
}

__global__ void k_scan_part3(const float* __restrict__ dbl, const __bf16* __restrict__ xi,
                             const float* __restrict__ xz, const float* __restrict__ dtw,
                             const float* __restrict__ dtb, const float* __restrict__ alog,
                             const float* __restrict__ dpar, const float* __restrict__ Hst,
                             __bf16* __restrict__ y) {
  int bc = blockIdx.x;
  int b = bc / kNC, c = bc % kNC;
  int d = threadIdx.x;
  float wdt[kDTR];
  #pragma unroll
  for (int e = 0; e < kDTR; e++) wdt[e] = dtw[d * kDTR + e];
  float Anr[kNS];
  #pragma unroll
  for (int n = 0; n < kNS; n++) Anr[n] = -__expf(alog[d * kNS + n]);
  float bdt = dtb[d], dp = dpar[d];
  float h[kNS];
  {
    const float* Hp = Hst + ((size_t)bc * kDI + d) * kNS;
    #pragma unroll
    for (int n = 0; n < kNS; n++) h[n] = Hp[n];
  }
  size_t row0 = (size_t)b * kL + (size_t)c * kCL;
  for (int l = 0; l < kCL; l++) {
    size_t row = row0 + l;
    if (l + kPFD < kCL) {
      __builtin_prefetch(dbl + (row + kPFD) * kXP, 0, 3);
      __builtin_prefetch(xi + (row + kPFD) * kDI + d, 0, 3);
      __builtin_prefetch(xz + (row + kPFD) * (2 * kDI) + kDI + d, 0, 3);
    }
    const f4* q = (const f4*)(dbl + row * kXP);
    f4 r0 = q[0], r1 = q[1];
    float dl = bdt;
    #pragma unroll
    for (int e = 0; e < 4; e++) { dl += r0[e] * wdt[e]; dl += r1[e] * wdt[4 + e]; }
    float dt = softplusf(dl);
    float xt = (float)xi[row * kDI + d];
    float dtxt = dt * xt;
    float ys = 0.f;
    #pragma unroll
    for (int n = 0; n < kNS; n++) {
      float Bn = q[2 + (n >> 2)][n & 3];           // dbl cols 8..23
      float Cn = q[6 + (n >> 2)][n & 3];           // dbl cols 24..39
      float dA = __expf(dt * Anr[n]);
      h[n] = dA * h[n] + dtxt * Bn;
      ys += h[n] * Cn;
    }
    float zv = xz[row * (2 * kDI) + kDI + d];
    float yo = (ys + dp * xt) * (zv * sigf(zv));
    y[row * kDI + d] = (__bf16)yo;
  }
}

// ---------- final layernorm + mean over L -> fe[16,128] ----------
__global__ void k_fnorm_mean(const float* __restrict__ x, const float* __restrict__ w,
                             const float* __restrict__ b, float* __restrict__ fe) {
  __shared__ float sm[8 * kDM];
  int bi = blockIdx.x;
  int wave = threadIdx.x >> 5, lane = threadIdx.x & 31;
  f4 acc = {0.f, 0.f, 0.f, 0.f};
  for (int l = wave; l < kL; l += 8) {
    f4 v = ((const f4*)(x + ((size_t)bi * kL + l) * kDM))[lane];
    float s  = v[0] + v[1] + v[2] + v[3];
    float s2 = v[0]*v[0] + v[1]*v[1] + v[2]*v[2] + v[3]*v[3];
    #pragma unroll
    for (int o = 16; o > 0; o >>= 1) { s += __shfl_xor(s, o); s2 += __shfl_xor(s2, o); }
    float m   = s * (1.f / kDM);
    float inv = rsqrtf(s2 * (1.f / kDM) - m * m + 1e-5f);
    #pragma unroll
    for (int e = 0; e < 4; e++) acc[e] += (v[e] - m) * inv;
  }
  #pragma unroll
  for (int e = 0; e < 4; e++) sm[wave * kDM + lane * 4 + e] = acc[e];
  __syncthreads();
  if (threadIdx.x < kDM) {
    float t = 0.f;
    #pragma unroll
    for (int wv = 0; wv < 8; wv++) t += sm[wv * kDM + threadIdx.x];
    fe[bi * kDM + threadIdx.x] = (t * (1.f / kL)) * w[threadIdx.x] + b[threadIdx.x];
  }
}

// ---------- LSTM head over T=16 steps + classifier ----------
__global__ void k_lstm(const float* __restrict__ fe, const int* __restrict__ actions,
                       const float* __restrict__ aemb, const float* __restrict__ wih,
                       const float* __restrict__ whh, const float* __restrict__ bih,
                       const float* __restrict__ bhh, const float* __restrict__ clsw,
                       const float* __restrict__ clsb, float* __restrict__ out) {
  __shared__ float xv[kDM + kAEMB];
  __shared__ float hv[kHID];
  __shared__ float gb[4 * kHID];
  int tid = threadIdx.x;           // 0..1023
  float c = 0.f;
  if (tid < kHID) hv[tid] = 0.f;
  __syncthreads();
  for (int t = 0; t < kT; t++) {
    if (tid < kDM) xv[tid] = fe[t * kDM + tid];
    else if (tid < kDM + kAEMB) {
      int prev = (t == 0) ? kNA : actions[t - 1];
      xv[tid] = aemb[prev * kAEMB + (tid - kDM)];
    }
    __syncthreads();
    float g = bih[tid] + bhh[tid];
    const float* wi = wih + (size_t)tid * (kDM + kAEMB);
    #pragma unroll 8
    for (int k = 0; k < kDM + kAEMB; k++) g += wi[k] * xv[k];
    const float* wh = whh + (size_t)tid * kHID;
    #pragma unroll 8
    for (int k = 0; k < kHID; k++) g += wh[k] * hv[k];
    gb[tid] = g;
    __syncthreads();
    if (tid < kHID) {
      float ig = sigf(gb[tid]);
      float fg = sigf(gb[kHID + tid]);
      float gg = tanhf(gb[2 * kHID + tid]);
      float og = sigf(gb[3 * kHID + tid]);
      c = fg * c + ig * gg;
      hv[tid] = og * tanhf(c);
    }
    __syncthreads();
    if (tid < kNA) {
      float a = clsb[tid];
      const float* cw = clsw + tid * kHID;
      for (int k = 0; k < kHID; k++) a += cw[k] * hv[k];
      out[t * kNA + tid] = a;
    }
    __syncthreads();
  }
}

// ---------- host orchestration ----------
extern "C" void kernel_launch(void* const* d_in, const int* in_sizes, int n_in,
                              void* d_out, int out_size, void* d_ws, size_t ws_size,
                              hipStream_t stream) {
  (void)in_sizes; (void)n_in; (void)out_size; (void)ws_size;
  const int*   frames = (const int*)  d_in[0];
  const int*   actions= (const int*)  d_in[1];
  const float* emb    = (const float*)d_in[2];
  const float* pos    = (const float*)d_in[3];
  const float* norm_w = (const float*)d_in[4];
  const float* norm_b = (const float*)d_in[5];
  const float* inw    = (const float*)d_in[6];
  const float* convw  = (const float*)d_in[7];
  const float* convb  = (const float*)d_in[8];
  const float* xpw    = (const float*)d_in[9];
  const float* dtw    = (const float*)d_in[10];
  const float* dtb    = (const float*)d_in[11];
  const float* alog   = (const float*)d_in[12];
  const float* dpar   = (const float*)d_in[13];
  const float* outw   = (const float*)d_in[14];
  const float* fnw    = (const float*)d_in[15];
  const float* fnb    = (const float*)d_in[16];
  const float* aemb   = (const float*)d_in[17];
  const float* wih    = (const float*)d_in[18];
  const float* whh    = (const float*)d_in[19];
  const float* bih    = (const float*)d_in[20];
  const float* bhh    = (const float*)d_in[21];
  const float* clsw   = (const float*)d_in[22];
  const float* clsb   = (const float*)d_in[23];

  char* ws = (char*)d_ws;
  size_t off = 0;
  auto alloc = [&](size_t bytes) { size_t o = off; off += (bytes + 255) & ~(size_t)255; return o; };
  float*  x     = (float*) (ws + alloc((size_t)kBL * kDM * 4));       // residual stream (f32)
  __bf16* lnbf  = (__bf16*)(ws + alloc((size_t)kBL * kDM * 2));       // layernorm out (bf16 A)
  float*  xz    = (float*) (ws + alloc((size_t)kBL * 2 * kDI * 4));   // in_proj out [BL,512]
  __bf16* xibf  = (__bf16*)(ws + alloc((size_t)kBL * kDI * 2));       // conv+silu out
  float*  dbl   = (float*) (ws + alloc((size_t)kBL * kXP * 4));       // x_proj out [BL,48]
  __bf16* ybf   = (__bf16*)(ws + alloc((size_t)kBL * kDI * 2));       // scan out
  float*  fe    = (float*) (ws + alloc((size_t)kT * kDM * 4));        // frame embeddings
  float*  Pbuf  = (float*) (ws + alloc((size_t)kT * kNC * kDI * kNS * 4)); // chunk dA products
  float*  Sbuf  = (float*) (ws + alloc((size_t)kT * kNC * kDI * kNS * 4)); // chunk partial states
  float*  Hst   = (float*) (ws + alloc((size_t)kT * kNC * kDI * kNS * 4)); // chunk start states
  __bf16* inwb  = (__bf16*)(ws + alloc((size_t)kNL * 2 * kDI * kDM * 2));
  __bf16* xpwb  = (__bf16*)(ws + alloc((size_t)kNL * kXP * kDI * 2));
  __bf16* outwb = (__bf16*)(ws + alloc((size_t)kNL * kDM * kDI * 2));

  // weights -> bf16 (cheap; deterministic every call)
  k_f2bf<<<(kNL * 2 * kDI * kDM + 255) / 256, 256, 0, stream>>>(inw, inwb, kNL * 2 * kDI * kDM);
  k_xpw_pad<<<(kNL * kXP * kDI + 255) / 256, 256, 0, stream>>>(xpw, xpwb);
  k_f2bf<<<(kNL * kDM * kDI + 255) / 256, 256, 0, stream>>>(outw, outwb, kNL * kDM * kDI);

  k_embed<<<kBL, kDM, 0, stream>>>(frames, emb, pos, x);

  const int Mtiles = kBL / 16;  // 4800
  for (int i = 0; i < kNL; i++) {
    k_layernorm_bf16<<<kBL / 8, 256, 0, stream>>>(x, norm_w + i * kDM, norm_b + i * kDM, lnbf);
    // in_proj: N=512, K=128, NT=4 -> 8 N-groups
    k_gemm_bf16<4, kDM, false><<<Mtiles * 8 / 8, 256, 0, stream>>>(
        lnbf, inwb + (size_t)i * 2 * kDI * kDM, xz, Mtiles, 8, 2 * kDI);
    k_conv_silu<<<kBL * kDI / 256, 256, 0, stream>>>(
        xz, convw + i * kDI * kKC, convb + i * kDI, xibf);
    // x_proj: N=48 (padded), K=256, NT=3 -> 1 group
    k_gemm_bf16<3, kDI, false><<<Mtiles / 8, 256, 0, stream>>>(
        xibf, xpwb + (size_t)i * kXP * kDI, dbl, Mtiles, 1, kXP);
    // chunked parallel selective scan
    k_scan_part1<<<kT * kNC, kDI, 0, stream>>>(
        dbl, xibf, dtw + i * kDI * kDTR, dtb + i * kDI, alog + i * kDI * kNS, Pbuf, Sbuf);
    k_scan_part2<<<kT, kDI, 0, stream>>>(Pbuf, Sbuf, Hst);
    k_scan_part3<<<kT * kNC, kDI, 0, stream>>>(
        dbl, xibf, xz, dtw + i * kDI * kDTR, dtb + i * kDI,
        alog + i * kDI * kNS, dpar + i * kDI, Hst, ybf);
    // out_proj fused with residual add: N=128, K=256, NT=4 -> 2 groups, ACC=true
    k_gemm_bf16<4, kDI, true><<<Mtiles * 2 / 8, 256, 0, stream>>>(
        ybf, outwb + (size_t)i * kDM * kDI, x, Mtiles, 2, kDM);
  }

  k_fnorm_mean<<<kT, 256, 0, stream>>>(x, fnw, fnb, fe);
  k_lstm<<<1, 1024, 0, stream>>>(fe, actions, aemb, wih, whh, bih, bhh, clsw, clsb,
                                 (float*)d_out);
}
